// EnhancedMambaFusion_40329742909520
// MI455X (gfx1250) — matrix operand order repaired and checked by reading, not compiled, not run on hardware
//
#include <hip/hip_runtime.h>
#include <hip/hip_bf16.h>
#include <math.h>

// CDNA5 wave32 WMMA types
typedef __attribute__((ext_vector_type(16))) _Float16 v16h;
typedef __attribute__((ext_vector_type(8)))  float    v8f;

// ---------------------------------------------------------------------------
// Generic GEMM: C[M,N] = act(A[M,K] @ W[N,K]^T + bias[N])
//   fp32 inputs converted on the fly to f16; fp32 accumulate via
//   v_wmma_f32_16x16x32_f16. Block = 128 threads (4 waves).
//   Block tile = 128M x 64N; each wave owns a 32M x 64N strip:
//   2 M-subtiles x 4 N-subtiles = 8 accumulators; each B fragment is
//   reused by 2 WMMAs (halves weight traffic vs 16M strips).
//   Requires: M % 128 == 0, K % 32 == 0, N % 16 == 0.
//   Out-of-range N tiles: W pointer clamped to last valid strip (in-bounds
//   loads, results discarded at store) -> zero control flow in the K loop.
//   act: 0 = none, 1 = relu, 2 = softplus
// ---------------------------------------------------------------------------
__device__ __forceinline__ void cvt4(v16h& d, int o, float4 q) {
  d[o + 0] = (_Float16)q.x; d[o + 1] = (_Float16)q.y;
  d[o + 2] = (_Float16)q.z; d[o + 3] = (_Float16)q.w;
}

__global__ __launch_bounds__(128) void gemm_f16_wmma(
    const float* __restrict__ A, int ldA,
    const float* __restrict__ W, int K,
    const float* __restrict__ bias,
    float* __restrict__ C, int ldC,
    int N, int act)
{
  const int lane = threadIdx.x & 31;
  const int wave = threadIdx.x >> 5;
  const int lm   = lane & 15;   // row (A) / col (B) within 16
  const int kg   = lane >> 4;   // K-group select (0/1)
  const int m0   = blockIdx.y * 128 + wave * 32;
  const int n0   = blockIdx.x * 64;

  v8f acc[2][4] = {};

  const float* __restrict__ arow0 = A + (size_t)(m0 + lm) * (size_t)ldA;
  const float* __restrict__ arow1 = arow0 + (size_t)16 * (size_t)ldA;

  // Precompute per-subtile W row pointers; clamp OOB tiles (loop stays branch-free)
  const float* wbase[4];
  bool nvalid[4];
#pragma unroll
  for (int nt = 0; nt < 4; ++nt) {
    int nb = n0 + nt * 16;
    nvalid[nt] = (nb < N);
    if (nb >= N) nb = N - 16;
    wbase[nt] = W + (size_t)(nb + lm) * (size_t)K;
  }

  for (int kk = 0; kk < K; kk += 32) {
    // A fragments (16x32 f16 each): lane lm holds row;
    // halves 0..7 -> K = kk+kg*8.. ; halves 8..15 -> K = kk+16+kg*8..
    v16h a0, a1;
    {
      const float4* p0 = (const float4*)(arow0 + kk + kg * 8);
      const float4* p1 = (const float4*)(arow0 + kk + 16 + kg * 8);
      cvt4(a0, 0, p0[0]); cvt4(a0, 4, p0[1]);
      cvt4(a0, 8, p1[0]); cvt4(a0, 12, p1[1]);
      const float4* q0 = (const float4*)(arow1 + kk + kg * 8);
      const float4* q1 = (const float4*)(arow1 + kk + 16 + kg * 8);
      cvt4(a1, 0, q0[0]); cvt4(a1, 4, q0[1]);
      cvt4(a1, 8, q1[0]); cvt4(a1, 12, q1[1]);
    }
#pragma unroll
    for (int nt = 0; nt < 4; ++nt) {
      // B fragment (32x16 f16): lane lm holds col; halves 0..7 -> K = kk+kg*8.. ; halves 8..15 -> K = kk+16+kg*8..
      v16h bf;
      const float4* wp = (const float4*)(wbase[nt] + kk + kg * 8);
      cvt4(bf, 0, wp[0]); cvt4(bf, 4, wp[1]);
      cvt4(bf, 8, wp[4]); cvt4(bf, 12, wp[5]);
      acc[0][nt] = __builtin_amdgcn_wmma_f32_16x16x32_f16(
          false, a0, false, bf, (short)0, acc[0][nt], false, false);
      acc[1][nt] = __builtin_amdgcn_wmma_f32_16x16x32_f16(
          false, a1, false, bf, (short)0, acc[1][nt], false, false);
    }
  }

  // Epilogue: C/D layout -> lane L: col = L%16, VGPR v -> row = v + 8*(L/16)
  const int r0 = m0 + (kg << 3);
#pragma unroll
  for (int ms = 0; ms < 2; ++ms) {
#pragma unroll
    for (int nt = 0; nt < 4; ++nt) {
      if (!nvalid[nt]) continue;   // wave-uniform
      const int n = n0 + nt * 16 + lm;
      const float bv = bias ? bias[n] : 0.0f;
#pragma unroll
      for (int v = 0; v < 8; ++v) {
        float val = acc[ms][nt][v] + bv;
        if (act == 1)      val = fmaxf(val, 0.0f);
        else if (act == 2) val = (val > 20.0f) ? val : log1pf(expf(val));
        C[(size_t)(r0 + ms * 16 + v) * (size_t)ldC + n] = val;
      }
    }
  }
}

// ---------------------------------------------------------------------------
// Row-wise LayerNorm (optionally fused residual add). In-place safe.
// ---------------------------------------------------------------------------
__global__ __launch_bounds__(256) void ln_kernel(
    const float* __restrict__ x, const float* __restrict__ res,
    const float* __restrict__ g, const float* __restrict__ b,
    float* __restrict__ y, int N)
{
  const int row = blockIdx.x;
  const float* xr = x + (size_t)row * N;
  const float* rr = res ? res + (size_t)row * N : nullptr;
  __shared__ float s1[256], s2[256];
  float sum = 0.f, sq = 0.f;
  for (int c = threadIdx.x; c < N; c += 256) {
    float v = xr[c]; if (rr) v += rr[c];
    sum += v; sq += v * v;
  }
  s1[threadIdx.x] = sum; s2[threadIdx.x] = sq;
  __syncthreads();
  for (int off = 128; off > 0; off >>= 1) {
    if (threadIdx.x < off) {
      s1[threadIdx.x] += s1[threadIdx.x + off];
      s2[threadIdx.x] += s2[threadIdx.x + off];
    }
    __syncthreads();
  }
  const float mean = s1[0] / (float)N;
  const float var  = s2[0] / (float)N - mean * mean;
  const float rs   = rsqrtf(var + 1e-5f);
  for (int c = threadIdx.x; c < N; c += 256) {
    float v = xr[c]; if (rr) v += rr[c];
    y[(size_t)row * N + c] = (v - mean) * rs * g[c] + b[c];
  }
}

// cat = [img, txt]  (1024 x 1024)
__global__ __launch_bounds__(256) void concat2_kernel(
    const float* __restrict__ a, const float* __restrict__ t, float* __restrict__ dst)
{
  const int i = blockIdx.x * 256 + threadIdx.x;
  const int row = i >> 10, col = i & 1023;
  dst[i] = (col < 512) ? a[row * 512 + col] : t[row * 512 + col - 512];
}

// gate logits (N=2, K=256) + softmax; writes GATE scratch and output section
__global__ __launch_bounds__(256) void gate_softmax_kernel(
    const float* __restrict__ H2, const float* __restrict__ w3,
    const float* __restrict__ b3, float* __restrict__ gate,
    float* __restrict__ out_gate)
{
  const int row = blockIdx.x * 256 + threadIdx.x;
  if (row >= 1024) return;
  const float* h = H2 + (size_t)row * 256;
  float l0 = b3[0], l1 = b3[1];
  for (int k = 0; k < 256; ++k) {
    const float hv = h[k];
    l0 += hv * w3[k];
    l1 += hv * w3[256 + k];
  }
  const float mx = fmaxf(l0, l1);
  const float e0 = expf(l0 - mx), e1 = expf(l1 - mx);
  const float inv = 1.0f / (e0 + e1);
  const float g0 = e0 * inv, g1 = e1 * inv;
  gate[row * 2 + 0] = g0; gate[row * 2 + 1] = g1;
  out_gate[row * 2 + 0] = g0; out_gate[row * 2 + 1] = g1;
}

// x = [gate0*img, gate1*txt]  (1024 x 1024)
__global__ __launch_bounds__(256) void build_x_kernel(
    const float* __restrict__ img, const float* __restrict__ txt,
    const float* __restrict__ gate, float* __restrict__ x)
{
  const int i = blockIdx.x * 256 + threadIdx.x;
  const int row = i >> 10, col = i & 1023;
  x[i] = (col < 512) ? gate[row * 2] * img[row * 512 + col]
                     : gate[row * 2 + 1] * txt[row * 512 + col - 512];
}

// L==1 causal conv collapses to last tap: xc = silu(xz[:, :2048]*cw[:,3] + cb)
__global__ __launch_bounds__(256) void conv_silu_kernel(
    const float* __restrict__ xz, const float* __restrict__ cw,
    const float* __restrict__ cb, float* __restrict__ xc)
{
  const int i = blockIdx.x * 256 + threadIdx.x;   // 1024*2048
  const int row = i >> 11, d = i & 2047;
  const float v = xz[(size_t)row * 4096 + d] * cw[d * 4 + 3] + cb[d];
  xc[i] = v / (1.0f + expf(-v));
}

// bdot[b] = sum_s B[b,s]*C[b,s]   (dbc cols 64..79 and 80..95)
__global__ __launch_bounds__(256) void bdot_kernel(
    const float* __restrict__ dbc, float* __restrict__ bdot)
{
  const int row = blockIdx.x * 256 + threadIdx.x;
  if (row >= 1024) return;
  const float* p = dbc + (size_t)row * 96;
  float s = 0.f;
#pragma unroll
  for (int k = 0; k < 16; ++k) s += p[64 + k] * p[80 + k];
  bdot[row] = s;
}

// Collapsed SSM (h0==0, L==1): u = (delta*xc*bdot + xc*D) * silu(z), in-place over xc
__global__ __launch_bounds__(256) void mamba_elem_kernel(
    float* __restrict__ xc, const float* __restrict__ delta,
    const float* __restrict__ xz, const float* __restrict__ bdot,
    const float* __restrict__ Dp)
{
  const int i = blockIdx.x * 256 + threadIdx.x;   // 1024*2048
  const int row = i >> 11, d = i & 2047;
  const float xcv = xc[i];
  const float y = delta[i] * xcv * bdot[row] + xcv * Dp[d];
  const float z = xz[(size_t)row * 4096 + 2048 + d];
  xc[i] = y * (z / (1.0f + expf(-z)));
}

// ---------------------------------------------------------------------------
extern "C" void kernel_launch(void* const* d_in, const int* in_sizes, int n_in,
                              void* d_out, int out_size, void* d_ws, size_t ws_size,
                              hipStream_t stream) {
  (void)in_sizes; (void)n_in; (void)out_size; (void)ws_size;

  const float* image = (const float*)d_in[0];
  const float* text  = (const float*)d_in[1];
  const float* img_w = (const float*)d_in[2];
  const float* img_b = (const float*)d_in[3];
  const float* img_g = (const float*)d_in[4];
  const float* img_bb= (const float*)d_in[5];
  const float* txt_w = (const float*)d_in[6];
  const float* txt_b = (const float*)d_in[7];
  const float* txt_g = (const float*)d_in[8];
  const float* txt_bb= (const float*)d_in[9];
  const float* gw1 = (const float*)d_in[10];
  const float* gb1 = (const float*)d_in[11];
  const float* gw2 = (const float*)d_in[12];
  const float* gb2 = (const float*)d_in[13];
  const float* gw3 = (const float*)d_in[14];
  const float* gb3 = (const float*)d_in[15];
  const float* ipw = (const float*)d_in[16];
  const float* cw  = (const float*)d_in[17];
  const float* cb  = (const float*)d_in[18];
  const float* xpw = (const float*)d_in[19];
  const float* dtw = (const float*)d_in[20];
  const float* dtb = (const float*)d_in[21];
  /* d_in[22] = A_log: unused (h0==0 and L==1 -> dA never multiplies state) */
  const float* Dp  = (const float*)d_in[23];
  const float* opw = (const float*)d_in[24];
  const float* mg  = (const float*)d_in[25];
  const float* mb  = (const float*)d_in[26];
  const float* fcw = (const float*)d_in[27];
  const float* fcb = (const float*)d_in[28];
  const float* fg  = (const float*)d_in[29];
  const float* fb  = (const float*)d_in[30];

  float* ws = (float*)d_ws;
  // Workspace layout (floats); overlapping regions are phase-disjoint.
  float* IMG_A = ws + 0;          // 1024x512   (align phase)
  float* IMG_B = ws + 524288;     // 1024x512
  float* TXT_A = ws + 1048576;    // 1024x512
  float* TXT_B = ws + 1572864;    // 1024x512
  float* XC    = ws + 0;          // 1024x2048  (mamba; reuses align buffers)
  float* CAT   = ws + 2097152;    // 1024x1024  (gate phase)
  float* H1    = ws + 3145728;    // 1024x512
  float* H2    = ws + 3670016;    // 1024x256
  float* DELTA = ws + 2097152;    // 1024x2048  (mamba; reuses CAT/H1/H2)
  float* FCOUT = ws + 2097152;    // 1024x256   (final; reuses DELTA)
  float* X     = ws + 4194304;    // 1024x1024
  float* O     = ws + 5242880;    // 1024x1024
  float* XZ    = ws + 6291456;    // 1024x4096
  float* GATE  = ws + 10485760;   // 1024x2
  float* BDOT  = ws + 10487808;   // 1024
  float* DBC   = ws + 10488832;   // 1024x96
  // total: 10,587,136 floats ~= 42.4 MB

  float* out_fused = (float*)d_out;            // [1024,256]
  float* out_gate  = (float*)d_out + 262144;   // [1024,2]

  const dim3 blk(128);
  auto gemm = [&](const float* A, int ldA, const float* W, int K,
                  const float* bias, float* C, int ldC, int N, int act) {
    dim3 grid((unsigned)((N + 63) / 64), 8 /* 1024/128 */);
    hipLaunchKernelGGL(gemm_f16_wmma, grid, blk, 0, stream,
                       A, ldA, W, K, bias, C, ldC, N, act);
  };

  // ---- align towers: 3x (GEMM + bias + relu, then LN in place) ----
  gemm(image, 512, img_w + 0 * 262144, 512, img_b + 0,    IMG_A, 512, 512, 1);
  ln_kernel<<<1024, 256, 0, stream>>>(IMG_A, nullptr, img_g + 0,    img_bb + 0,    IMG_A, 512);
  gemm(IMG_A, 512, img_w + 1 * 262144, 512, img_b + 512,  IMG_B, 512, 512, 1);
  ln_kernel<<<1024, 256, 0, stream>>>(IMG_B, nullptr, img_g + 512,  img_bb + 512,  IMG_B, 512);
  gemm(IMG_B, 512, img_w + 2 * 262144, 512, img_b + 1024, IMG_A, 512, 512, 1);
  ln_kernel<<<1024, 256, 0, stream>>>(IMG_A, nullptr, img_g + 1024, img_bb + 1024, IMG_A, 512);

  gemm(text, 512, txt_w + 0 * 262144, 512, txt_b + 0,    TXT_A, 512, 512, 1);
  ln_kernel<<<1024, 256, 0, stream>>>(TXT_A, nullptr, txt_g + 0,    txt_bb + 0,    TXT_A, 512);
  gemm(TXT_A, 512, txt_w + 1 * 262144, 512, txt_b + 512,  TXT_B, 512, 512, 1);
  ln_kernel<<<1024, 256, 0, stream>>>(TXT_B, nullptr, txt_g + 512,  txt_bb + 512,  TXT_B, 512);
  gemm(TXT_B, 512, txt_w + 2 * 262144, 512, txt_b + 1024, TXT_A, 512, 512, 1);
  ln_kernel<<<1024, 256, 0, stream>>>(TXT_A, nullptr, txt_g + 1024, txt_bb + 1024, TXT_A, 512);

  // ---- gate MLP + softmax ----
  concat2_kernel<<<4096, 256, 0, stream>>>(IMG_A, TXT_A, CAT);
  gemm(CAT, 1024, gw1, 1024, gb1, H1, 512, 512, 1);
  gemm(H1,  512,  gw2, 512,  gb2, H2, 256, 256, 1);
  gate_softmax_kernel<<<4, 256, 0, stream>>>(H2, gw3, gb3, GATE, out_gate);
  build_x_kernel<<<4096, 256, 0, stream>>>(IMG_A, TXT_A, GATE, X);

  // ---- 3 mamba layers (L==1 collapsed SSM) ----
  for (int i = 0; i < 3; ++i) {
    const float* ipw_i = ipw + (size_t)i * 4096 * 1024;
    const float* cw_i  = cw  + (size_t)i * 2048 * 4;
    const float* cb_i  = cb  + (size_t)i * 2048;
    const float* xpw_i = xpw + (size_t)i * 96 * 2048;
    const float* dtw_i = dtw + (size_t)i * 2048 * 64;
    const float* dtb_i = dtb + (size_t)i * 2048;
    const float* Dp_i  = Dp  + (size_t)i * 2048;
    const float* opw_i = opw + (size_t)i * 1024 * 2048;
    const float* mg_i  = mg  + (size_t)i * 1024;
    const float* mb_i  = mb  + (size_t)i * 1024;

    gemm(X, 1024, ipw_i, 1024, nullptr, XZ, 4096, 4096, 0);               // in_proj
    conv_silu_kernel<<<8192, 256, 0, stream>>>(XZ, cw_i, cb_i, XC);       // conv tap + silu
    gemm(XC, 2048, xpw_i, 2048, nullptr, DBC, 96, 96, 0);                 // x_proj
    bdot_kernel<<<4, 256, 0, stream>>>(DBC, BDOT);                        // B.C
    gemm(DBC, 96, dtw_i, 64, dtb_i, DELTA, 2048, 2048, 2);                // dt_proj+softplus
    mamba_elem_kernel<<<8192, 256, 0, stream>>>(XC, DELTA, XZ, BDOT, Dp_i); // u (in-place)
    gemm(XC, 2048, opw_i, 2048, nullptr, O, 1024, 1024, 0);               // out_proj
    ln_kernel<<<1024, 256, 0, stream>>>(O, X, mg_i, mb_i, X, 1024);       // LN(res + o)
  }

  // ---- final fc + LN -> d_out ----
  gemm(X, 1024, fcw, 1024, fcb, FCOUT, 256, 256, 0);
  ln_kernel<<<1024, 256, 0, stream>>>(FCOUT, nullptr, fg, fb, out_fused, 256);
}
